// Attention_10213432230262
// MI455X (gfx1250) — compile-verified, matrix operand order checked by reading, not run on hardware
//
#include <hip/hip_runtime.h>
#include <math.h>

// Problem constants (from the reference)
#define Hdim 1024
#define Sdim 2048
#define Bdim 32
#define K2   2048   // 2*H (GEMM K over enc_output part)
#define K3   3072   // 3*H (W_w row stride)
#define MTOK (Bdim * Sdim)

typedef __attribute__((ext_vector_type(16))) __bf16 v16bf;
typedef __attribute__((ext_vector_type(8)))  float  v8f;

union Frag { uint4 q[2]; v16bf v; };

// round-to-nearest-even f32 -> bf16, pack two into a dword
__device__ __forceinline__ unsigned pack_bf16x2(float lo, float hi) {
    union { float f; unsigned u; } a, b;
    a.f = lo; b.f = hi;
    unsigned ra = (a.u + 0x7FFFu + ((a.u >> 16) & 1u)) >> 16;
    unsigned rb = (b.u + 0x7FFFu + ((b.u >> 16) & 1u)) >> 16;
    return (rb << 16) | (ra & 0xFFFFu);
}

// ---------------------------------------------------------------------------
// Kernel 0: W_w[:, :2H] f32 -> bf16 (row-major [n][k], k stride = K2)
// ---------------------------------------------------------------------------
__global__ void conv_w_kernel(const float* __restrict__ W,
                              unsigned short* __restrict__ w1bf) {
    int idx = blockIdx.x * 256 + threadIdx.x;   // each thread: 2 elements
    int i2  = idx * 2;
    int n   = i2 >> 11;            // / K2
    int k   = i2 & (K2 - 1);
    const float* src = W + (size_t)n * K3 + k;
    *(unsigned*)(w1bf + i2) = pack_bf16x2(src[0], src[1]);
}

// ---------------------------------------------------------------------------
// Kernel 1a: partial[jc][b][n] = sum_{j in chunk jc} dec[b][j] * W_w[n][2H+j]
// W2 is read exactly once total; all 32 batches accumulated per thread.
// ---------------------------------------------------------------------------
__global__ void pre_part_kernel(const float* __restrict__ W,
                                const float* __restrict__ dec,
                                float* __restrict__ part) {
    const int n  = blockIdx.x * 256 + threadIdx.x;
    const int jc = blockIdx.y;                       // K chunk of 256
    const float* wrow = W + (size_t)n * K3 + K2 + jc * 256;
    const float* dj   = dec + jc * 256;
    float acc[Bdim];
    #pragma unroll
    for (int b = 0; b < Bdim; ++b) acc[b] = 0.f;
    for (int j = 0; j < 256; ++j) {
        float w = wrow[j];
        #pragma unroll
        for (int b = 0; b < Bdim; ++b)
            acc[b] += dj[b * Hdim + j] * w;
    }
    #pragma unroll
    for (int b = 0; b < Bdim; ++b)
        part[((size_t)jc * Bdim + b) * Hdim + n] = acc[b];
}

// ---------------------------------------------------------------------------
// Kernel 1b: pre[b][n] = W_b[n] + sum_jc partial[jc][b][n]
// ---------------------------------------------------------------------------
__global__ void pre_reduce_kernel(const float* __restrict__ part,
                                  const float* __restrict__ Wb,
                                  float* __restrict__ pre) {
    int idx = blockIdx.x * 256 + threadIdx.x;        // over B*H
    int n   = idx & (Hdim - 1);
    float s = Wb[n];
    #pragma unroll
    for (int jc = 0; jc < 4; ++jc)
        s += part[(size_t)jc * Bdim * Hdim + idx];
    pre[idx] = s;
}

// ---------------------------------------------------------------------------
// stage one 64 x 512 f32 chunk of enc into an LDS bf16 buffer (stride 520)
// ---------------------------------------------------------------------------
__device__ __forceinline__ void stage_chunk(const float* __restrict__ enc,
                                            unsigned short* __restrict__ dst,
                                            int token0, int kc, int tid) {
    #pragma unroll
    for (int it = 0; it < 16; ++it) {
        int f   = it * 512 + tid;   // 0..8191, 4 floats each
        int row = f >> 7;
        int kq  = (f & 127) << 2;
        const float4 g = *(const float4*)(enc + (size_t)(token0 + row) * K2
                                          + kc * 512 + kq);
        uint2 p;
        p.x = pack_bf16x2(g.x, g.y);
        p.y = pack_bf16x2(g.z, g.w);
        *(uint2*)(dst + row * 520 + kq) = p;
    }
}

// ---------------------------------------------------------------------------
// Kernel 2: fused scores GEMM
//   scores[t] = sum_n tanh( sum_k enc[t][k]*W[n][k] + pre[b][n] ) * Vw[n] + Vb
// Block: 64 tokens, 512 threads (16 waves). Wave = (mhalf = w>>3) x
// (nwave = w&7): owns 32 tokens (2 m-subtiles of 16) and 128 N columns
// (8 n-tiles of 16). A double-buffered in LDS in K-chunks of 512 so the
// staging (global_load + cvt + ds_store) of chunk kc+1 overlaps the WMMA
// stream of chunk kc. W (bf16, in L2) loaded per-lane in ISA fragment
// layout: two contiguous 16B chunks per lane.
// ---------------------------------------------------------------------------
__global__ __launch_bounds__(512)
void scores_kernel(const float* __restrict__ enc,
                   const unsigned short* __restrict__ w1bf,
                   const float* __restrict__ pre,
                   const float* __restrict__ Vw,
                   const float* __restrict__ Vb,
                   const float* __restrict__ mask,
                   float* __restrict__ scores) {
    __shared__ unsigned short A_sh[2][64 * 520];   // 2 x 66.5 KB
    __shared__ float sred[16][32];

    const int token0 = blockIdx.x * 64;
    const int b      = token0 >> 11;            // 64 | 2048 => one batch per block
    const int wave   = threadIdx.x >> 5;
    const int lane   = threadIdx.x & 31;
    const int nwave  = wave & 7;                // N group (128 cols each)
    const int mhalf  = wave >> 3;               // token half (32 tokens each)
    const int nn     = lane & 15;               // N (or M) index within half-wave
    const int c0     = (lane < 16) ? 0 : 8;     // K sub-offset per ISA 16-bit layout

    // accumulators: 8 n-tiles x 2 m-subtiles, init with per-(b,n) bias
    v8f acc[8][2];
    #pragma unroll
    for (int i = 0; i < 8; ++i) {
        float pv = pre[b * Hdim + nwave * 128 + i * 16 + nn];
        #pragma unroll
        for (int mi = 0; mi < 2; ++mi)
            #pragma unroll
            for (int r = 0; r < 8; ++r)
                acc[i][mi][r] = pv;
    }

    // prologue: stage chunk 0
    stage_chunk(enc, A_sh[0], token0, 0, threadIdx.x);
    __syncthreads();

    for (int kc = 0; kc < 4; ++kc) {
        const int cur = kc & 1;
        // overlap: stage next chunk into the other buffer (no barrier needed;
        // compute below only touches A_sh[cur])
        if (kc < 3)
            stage_chunk(enc, A_sh[cur ^ 1], token0, kc + 1, threadIdx.x);

        const unsigned short* Abuf = A_sh[cur];
        #pragma unroll
        for (int i = 0; i < 8; ++i) {
            const unsigned short* wrow =
                w1bf + (size_t)(nwave * 128 + i * 16 + nn) * K2 + kc * 512;
            #pragma unroll
            for (int ks = 0; ks < 16; ++ks) {
                const int kb = ks * 32;
                Frag bf;   // B fragment: row n, K {c0..c0+7} u {16+c0..16+c0+7}
                bf.q[0] = *(const uint4*)(wrow + kb + c0);
                bf.q[1] = *(const uint4*)(wrow + kb + c0 + 16);
                #pragma unroll
                for (int mi = 0; mi < 2; ++mi) {
                    Frag af;   // A fragment from LDS, M = mhalf*32 + mi*16 + nn
                    const unsigned short* arow =
                        Abuf + (mhalf * 32 + mi * 16 + nn) * 520 + kb + c0;
                    af.q[0] = *(const uint4*)(arow);
                    af.q[1] = *(const uint4*)(arow + 16);
                    acc[i][mi] = __builtin_amdgcn_wmma_f32_16x16x32_bf16(
                        false, af.v, false, bf.v, (short)0, acc[i][mi],
                        false, false);
                }
            }
        }
        // separates: this chunk's A reads from next overwrite of A_sh[cur],
        // and next chunk's staged writes from its compute reads
        __syncthreads();
    }

    // epilogue: tanh, scale by Vw, reduce over N (per-i sums first, then lanes)
    float sA[2][8];
    #pragma unroll
    for (int mi = 0; mi < 2; ++mi)
        #pragma unroll
        for (int r = 0; r < 8; ++r) sA[mi][r] = 0.f;

    #pragma unroll
    for (int i = 0; i < 8; ++i) {
        float vw = Vw[nwave * 128 + i * 16 + nn];
        #pragma unroll
        for (int mi = 0; mi < 2; ++mi)
            #pragma unroll
            for (int r = 0; r < 8; ++r)
                sA[mi][r] += tanhf(acc[i][mi][r]) * vw;
    }
    #pragma unroll
    for (int mi = 0; mi < 2; ++mi)
        #pragma unroll
        for (int r = 0; r < 8; ++r) {
            float v = sA[mi][r];
            v += __shfl_xor(v, 8, 32);
            v += __shfl_xor(v, 4, 32);
            v += __shfl_xor(v, 2, 32);
            v += __shfl_xor(v, 1, 32);
            sA[mi][r] = v;
        }
    // lanes 0/16 hold sums for M = r / M = r+8 of each m-subtile
    if (lane == 0 || lane == 16) {
        int hb = (lane >> 4) * 8;
        #pragma unroll
        for (int mi = 0; mi < 2; ++mi)
            #pragma unroll
            for (int r = 0; r < 8; ++r)
                sred[wave][mi * 16 + hb + r] = sA[mi][r];
    }
    __syncthreads();
    if (threadIdx.x < 64) {
        int t = threadIdx.x;
        int h = t >> 5;                 // which token half
        float sum = 0.f;
        #pragma unroll
        for (int w = 0; w < 8; ++w) sum += sred[h * 8 + w][t & 31];
        int tok = token0 + t;
        float s = sum + Vb[0];
        scores[tok] = (mask[tok] > 0.f) ? s : -__builtin_huge_valf();
    }
}

// ---------------------------------------------------------------------------
// Kernel 3: softmax over S per batch (in place)
// ---------------------------------------------------------------------------
__global__ void softmax_kernel(float* __restrict__ sc) {
    const int b = blockIdx.x;
    __shared__ float red[256];
    float* row = sc + b * Sdim;

    float m = -__builtin_huge_valf();
    for (int i = threadIdx.x; i < Sdim; i += 256) m = fmaxf(m, row[i]);
    red[threadIdx.x] = m;
    __syncthreads();
    for (int o = 128; o > 0; o >>= 1) {
        if (threadIdx.x < o)
            red[threadIdx.x] = fmaxf(red[threadIdx.x], red[threadIdx.x + o]);
        __syncthreads();
    }
    const float mx = red[0];
    __syncthreads();

    float s = 0.f;
    for (int i = threadIdx.x; i < Sdim; i += 256) {
        float e = __expf(row[i] - mx);
        row[i] = e;
        s += e;
    }
    red[threadIdx.x] = s;
    __syncthreads();
    for (int o = 128; o > 0; o >>= 1) {
        if (threadIdx.x < o) red[threadIdx.x] += red[threadIdx.x + o];
        __syncthreads();
    }
    const float inv = 1.f / red[0];
    for (int i = threadIdx.x; i < Sdim; i += 256) row[i] *= inv;
}

// ---------------------------------------------------------------------------
// Kernel 4: context[b][d] = sum_s attn[b][s] * enc[b][s][d]   (memory bound)
// ---------------------------------------------------------------------------
__global__ void context_kernel(const float* __restrict__ enc,
                               const float* __restrict__ attn,
                               float* __restrict__ out) {
    const int b = blockIdx.y;
    const int d = blockIdx.x * 256 + threadIdx.x;
    const float* e = enc + (size_t)b * Sdim * K2 + d;
    const float* a = attn + b * Sdim;
    float acc = 0.f;
    #pragma unroll 8
    for (int s = 0; s < Sdim; ++s)
        acc += a[s] * e[(size_t)s * K2];
    out[b * K2 + d] = acc;
}

// ---------------------------------------------------------------------------
extern "C" void kernel_launch(void* const* d_in, const int* in_sizes, int n_in,
                              void* d_out, int out_size, void* d_ws, size_t ws_size,
                              hipStream_t stream) {
    const float* enc  = (const float*)d_in[0];  // (B,S,2H)
    const float* mask = (const float*)d_in[1];  // (B,S,1)
    const float* dec  = (const float*)d_in[2];  // (1,B,H)
    const float* Ww   = (const float*)d_in[3];  // (H,3H)
    const float* Wb   = (const float*)d_in[4];  // (H,)
    const float* Vw   = (const float*)d_in[5];  // (1,H)
    const float* Vb   = (const float*)d_in[6];  // (1,)
    float* out = (float*)d_out;                 // (B,2H)

    char* ws = (char*)d_ws;
    size_t off = 0;
    unsigned short* w1bf = (unsigned short*)(ws + off); off += (size_t)Hdim * K2 * 2;      // 4 MB
    float* pre    = (float*)(ws + off); off += (size_t)Bdim * Hdim * 4;                    // 128 KB
    float* scores = (float*)(ws + off); off += (size_t)MTOK * 4;                           // 256 KB
    float* part   = (float*)(ws + off);                                                    // 512 KB

    conv_w_kernel<<<(Hdim * K2 / 2) / 256, 256, 0, stream>>>(Ww, w1bf);
    pre_part_kernel<<<dim3(Hdim / 256, 4), 256, 0, stream>>>(Ww, dec, part);
    pre_reduce_kernel<<<(Bdim * Hdim) / 256, 256, 0, stream>>>(part, Wb, pre);
    scores_kernel<<<MTOK / 64, 512, 0, stream>>>(enc, w1bf, pre, Vw, Vb, mask, scores);
    softmax_kernel<<<Bdim, 256, 0, stream>>>(scores);
    context_kernel<<<dim3(K2 / 256, Bdim), 256, 0, stream>>>(enc, scores, out);
}
